// FBSDE_72619307041130
// MI455X (gfx1250) — matrix operand order repaired and checked by reading, not compiled, not run on hardware
//
#include <hip/hip_runtime.h>
#include <cstdint>
#include <cstddef>

// ---------------- types ----------------
typedef __bf16 bf16_t;
typedef __attribute__((ext_vector_type(16))) __bf16 v16bf;
typedef __attribute__((ext_vector_type(8)))  __bf16 v8bf;
typedef __attribute__((ext_vector_type(8)))  float  v8f;

// ---------------- problem constants (from setup_inputs) ----------------
#define B_    1024
#define D_    8
#define NS_   100
#define LAG_  10
#define W_    11
#define R_    (B_*W_)        /* 11264 rows of (b,window) */
#define H_    512
#define G4_   2048           /* 4*H gates */
#define KSIG  4369           /* 1 + 16 + 256 + 4096 */
#define KPAD  4384           /* KSIG padded to multiple of 32 */
#define MU_F     0.05f
#define SIGMA_F  0.2f

__device__ __forceinline__ float sigm(float x){ return 1.0f/(1.0f+expf(-x)); }

// ================= SDE paths + windowed Brownian sums =================
__global__ void sde_kernel(const float* __restrict__ ts,
                           const float* __restrict__ x0,
                           const float* __restrict__ noise,
                           float* __restrict__ xs,        // B x 101 x 8
                           float* __restrict__ sum_inc)   // B x 11 x 8
{
    int g = blockIdx.x*blockDim.x + threadIdx.x;
    if (g >= B_*D_) return;
    int b = g >> 3, ch = g & 7;
    float x = x0[b*D_ + ch];
    xs[((size_t)b*(NS_+1))*D_ + ch] = x;
    float acc = 0.0f;
    for (int t = 0; t < NS_; ++t) {
        float h  = ts[t+1] - ts[t];
        float dw = noise[((size_t)b*NS_ + t)*D_ + ch] * sqrtf(h);
        x = x + MU_F*x*h + SIGMA_F*x*dw;
        xs[((size_t)b*(NS_+1) + t + 1)*D_ + ch] = x;
        acc += dw;
        if ((t % LAG_) == LAG_-1) {
            sum_inc[((size_t)b*W_ + t/LAG_)*D_ + ch] = acc;
            acc = 0.0f;
        }
    }
    sum_inc[((size_t)b*W_ + (W_-1))*D_ + ch] = 0.0f;
}

// ================= payoff =================
__global__ void payoff_kernel(const float* __restrict__ xs, float* __restrict__ pay)
{
    int b = blockIdx.x*blockDim.x + threadIdx.x;
    if (b >= B_) return;
    float mx = -3.4e38f, last = 0.0f;
    for (int t = 0; t <= NS_; ++t) {
        float s = 0.0f;
        #pragma unroll
        for (int ch = 0; ch < D_; ++ch) s += xs[((size_t)b*(NS_+1) + t)*D_ + ch];
        s *= (1.0f/D_);
        mx = fmaxf(mx, s);
        if (t == NS_) last = s;
    }
    pay[b] = mx - last;
}

// ================= depth-3 lead-lag signature, one block per (b,w) =====
// tx row layout: [tc, S1(16), S2(256), S3(4096), zero pad to 4384] in bf16
__global__ __launch_bounds__(256) void sig_kernel(const float* __restrict__ xs,
                                                  const float* __restrict__ ts,
                                                  bf16_t* __restrict__ tx)
{
    const int bw = blockIdx.x;            // 0..R_-1
    const int b = bw / W_, w = bw % W_;
    const int tid = threadIdx.x;          // 0..255
    const int i3 = tid >> 4, j3 = tid & 15;

    __shared__ float delta[LAG_][D_];     // q_{i+1} - q_i
    __shared__ float dx[16];

    if (tid < LAG_*D_) {
        int i = tid >> 3, ch = tid & 7;
        float qa, qb;
        if (w == 0) {
            float xv = xs[((size_t)b*(NS_+1))*D_ + ch];   // x0
            qa = (i == 0) ? 0.0f : xv;
            qb = xv;
        } else {
            int base = (w-1)*LAG_;
            qa = xs[((size_t)b*(NS_+1) + base + i    )*D_ + ch];
            qb = xs[((size_t)b*(NS_+1) + base + i + 1)*D_ + ch];
        }
        delta[i][ch] = qb - qa;
    }

    float s1 = 0.0f, s2 = 0.0f;
    float S3[16];
    #pragma unroll
    for (int k = 0; k < 16; ++k) S3[k] = 0.0f;

    for (int step = 0; step < 2*LAG_; ++step) {
        __syncthreads();
        if (tid < 16) {
            int i = step >> 1;
            bool even = ((step & 1) == 0);
            float v = delta[i][tid & 7];
            // even step: dx = (0, delta); odd step: dx = (delta, 0)
            dx[tid] = (((tid >= 8)) == even) ? v : 0.0f;
        }
        __syncthreads();
        float di = dx[i3], dj = dx[j3];
        // newS3[i,j,k] += dx_i dx_j dx_k /6 + S2[i,j] dx_k + S1[i] dx_j dx_k /2
        float coef = di*dj*(1.0f/6.0f) + s2 + s1*dj*0.5f;
        #pragma unroll
        for (int k = 0; k < 16; ++k) S3[k] += coef * dx[k];
        s2 += di*dj*0.5f + s1*dj;
        s1 += di;
    }
    __syncthreads();

    const size_t row = (size_t)bw * KPAD;
    if (tid == 0)            tx[row]           = (bf16_t)ts[w*LAG_];
    if ((tid & 15) == 0)     tx[row + 1 + i3]  = (bf16_t)s1;
    tx[row + 17 + tid] = (bf16_t)s2;
    #pragma unroll
    for (int k = 0; k < 16; ++k)
        tx[row + 273 + (size_t)tid*16 + k] = (bf16_t)S3[k];
    if (tid < KPAD - KSIG)   tx[row + KSIG + tid] = (bf16_t)0.0f;
}

// ================= f32 -> bf16 weight conversion (with column pad) =====
__global__ void cvt_pad_kernel(const float* __restrict__ src, bf16_t* __restrict__ dst,
                               int rows, int scols, int dcols)
{
    size_t idx = (size_t)blockIdx.x*blockDim.x + threadIdx.x;
    size_t tot = (size_t)rows*dcols;
    if (idx >= tot) return;
    int r = (int)(idx / dcols), c = (int)(idx % dcols);
    dst[idx] = (c < scols) ? (bf16_t)src[(size_t)r*scols + c] : (bf16_t)0.0f;
}

__global__ void bias_sum_kernel(const float* a, const float* b, float* o, int n)
{
    int i = blockIdx.x*blockDim.x + threadIdx.x;
    if (i < n) o[i] = a[i] + b[i];
}

// ================= WMMA bf16 GEMM:  C[M,N] = A[M,K] · W[N,K]^T =========
// block = 256 threads = 8 waves; block tile 128(M) x 64(N); wave tile 16x64
template<bool RELU, bool WF32, bool WBF16, bool BIAS>
__global__ __launch_bounds__(256) void gemm_bf16(
    const bf16_t* __restrict__ A, const bf16_t* __restrict__ Wt,
    const float* __restrict__ bias,
    float* __restrict__ Cf, bf16_t* __restrict__ Cb,
    int M, int N, int K)
{
    const int lane = threadIdx.x & 31;
    const int wv   = threadIdx.x >> 5;
    const int row0 = blockIdx.y*128 + wv*16;
    const int col0 = blockIdx.x*64;

    v8f acc[4] = {};

    // A fragment: 16x32 bf16. lane<16: M=lane, K {0..15}; lane>=16: K {16..31}
    const int ma = row0 + (lane & 15);
    const bf16_t* pa = A + (size_t)ma*K + ((lane >> 4) << 4);
    // B fragment from W (N x K, row-major): n = col0+t*16+(lane&15), 16 consecutive k
    const bf16_t* pb = Wt + (size_t)(col0 + (lane & 15))*K + ((lane >> 4) << 4);
    const size_t bstep = (size_t)16 * K;

    for (int k0 = 0; k0 < K; k0 += 32) {
        v8bf alo = *(const v8bf*)(pa + k0);
        v8bf ahi = *(const v8bf*)(pa + k0 + 8);
        v16bf a;
        #pragma unroll
        for (int i = 0; i < 8; ++i) { a[i] = alo[i]; a[8+i] = ahi[i]; }
        #pragma unroll
        for (int t = 0; t < 4; ++t) {
            v16bf b = *(const v16bf*)(pb + (size_t)t*bstep + k0);
            acc[t] = __builtin_amdgcn_wmma_f32_16x16x32_bf16(
                        false, a, false, b, (short)0, acc[t], false, false);
        }
    }

    // C layout: VGPR r -> row (row0 + (lane>=16 ? 8:0) + r), col = col0+t*16+(lane&15)
    const int mo = row0 + ((lane >> 4) << 3);
    const int nl = lane & 15;
    #pragma unroll
    for (int t = 0; t < 4; ++t) {
        const int nc = col0 + t*16 + nl;
        float bv = 0.0f;
        if (BIAS) bv = bias[nc];
        #pragma unroll
        for (int r = 0; r < 8; ++r) {
            float v = acc[t][r] + bv;
            if (RELU)  v = fmaxf(v, 0.0f);
            if (WF32)  Cf[(size_t)(mo + r)*N + nc] = v;
            if (WBF16) Cb[(size_t)(mo + r)*N + nc] = (bf16_t)v;
        }
    }
}

// ================= LSTM state init / cell ==============================
__global__ void init_hc_kernel(bf16_t* __restrict__ h, float* __restrict__ c)
{
    int g = blockIdx.x*blockDim.x + threadIdx.x;
    if (g >= B_*H_) return;
    h[g] = (bf16_t)0.0f;
    c[g] = 0.0f;
}

__global__ void lstm_cell_kernel(const float* __restrict__ Xproj,  // R_ x 2048 (bias folded)
                                 const float* __restrict__ hW,     // B_ x 2048
                                 float* __restrict__ c,            // B_ x 512
                                 bf16_t* __restrict__ h,           // B_ x 512
                                 bf16_t* __restrict__ hs,          // R_ x 512
                                 int t)
{
    int g = blockIdx.x*blockDim.x + threadIdx.x;
    if (g >= B_*H_) return;
    int b = g >> 9, j = g & (H_-1);
    size_t xr = ((size_t)b*W_ + t)*G4_;
    size_t hr = (size_t)b*G4_;
    float gi = Xproj[xr         + j] + hW[hr         + j];
    float gf = Xproj[xr +   H_  + j] + hW[hr +   H_  + j];
    float gg = Xproj[xr + 2*H_  + j] + hW[hr + 2*H_  + j];
    float go = Xproj[xr + 3*H_  + j] + hW[hr + 3*H_  + j];
    float cs = sigm(gf)*c[g] + sigm(gi)*tanhf(gg);
    float hh = sigm(go)*tanhf(cs);
    c[g] = cs;
    bf16_t hb = (bf16_t)hh;
    h[g] = hb;
    hs[((size_t)b*W_ + t)*H_ + j] = hb;
}

// ================= small output head (O = 1 or 8) ======================
__global__ void head_kernel(const bf16_t* __restrict__ in,   // R_ x 512
                            const float* __restrict__ W2,    // O x 512
                            const float* __restrict__ b2,
                            float* __restrict__ out, int O)
{
    int g = blockIdx.x*blockDim.x + threadIdx.x;
    if (g >= R_*O) return;
    int r = g / O, o = g % O;
    const bf16_t* a = in + (size_t)r*H_;
    const float*  w = W2 + (size_t)o*H_;
    float s = 0.0f;
    #pragma unroll 8
    for (int k = 0; k < H_; ++k) s += (float)a[k]*w[k];
    out[(size_t)r*O + o] = s + b2[o];
}

// ================= deterministic loss reduction ========================
__global__ __launch_bounds__(1024) void loss_kernel(
    const float* __restrict__ Y,        // R_ (d_out+1)
    const float* __restrict__ Z,        // R_ x 8
    const float* __restrict__ sum_inc,  // B_ x 11 x 8
    const float* __restrict__ pay,      // B_
    const float* __restrict__ ts,
    float* __restrict__ out0)
{
    __shared__ float red[1024];
    int tid = threadIdx.x;
    float acc = 0.0f;
    for (int idx = tid; idx < R_; idx += 1024) {
        int b = idx / W_, t = idx % W_;
        float p = Y[idx];
        #pragma unroll
        for (int dd = 0; dd < D_; ++dd)
            p += Z[(size_t)idx*D_ + dd] * sum_inc[((size_t)b*W_ + t)*D_ + dd];
        float tgt;
        if (t < W_-1) {
            float disc = expf(-MU_F*(ts[t + LAG_] - ts[t*LAG_]));
            tgt = disc * Y[(size_t)b*W_ + t + 1];
        } else {
            tgt = pay[b];
        }
        float e = p - tgt;
        acc += e*e;
    }
    red[tid] = acc;
    __syncthreads();
    for (int s = 512; s > 0; s >>= 1) {
        if (tid < s) red[tid] += red[tid + s];
        __syncthreads();
    }
    if (tid == 0) out0[0] = red[0] * (1.0f/B_);
}

// =======================================================================
extern "C" void kernel_launch(void* const* d_in, const int* in_sizes, int n_in,
                              void* d_out, int out_size, void* d_ws, size_t ws_size,
                              hipStream_t stream)
{
    const float* ts    = (const float*)d_in[0];
    const float* x0    = (const float*)d_in[1];
    const float* noise = (const float*)d_in[2];
    // d_in[3] = lag (hardcoded LAG_=10)
    const float* P[2][10];
    for (int n = 0; n < 2; ++n)
        for (int j = 0; j < 10; ++j)
            P[n][j] = (const float*)d_in[4 + n*10 + j];
    // P[n]: 0 Wih, 1 Whh, 2 bih, 3 bhh, 4 W0, 5 b0, 6 W1, 7 b1, 8 W2, 9 b2

    float* out  = (float*)d_out;
    float* Ybuf = out + 1;
    float* pay  = out + 1 + R_;

    // ---- workspace carve-up ----
    size_t off = 0;
    auto carve = [&](size_t bytes) -> char* {
        char* p = (char*)d_ws + off;
        off += (bytes + 255) & ~(size_t)255;
        return p;
    };
    float*  xs     = (float*) carve((size_t)B_*(NS_+1)*D_*4);
    float*  sinc   = (float*) carve((size_t)B_*W_*D_*4);
    bf16_t* txpad  = (bf16_t*)carve((size_t)R_*KPAD*2);
    bf16_t* WihB   = (bf16_t*)carve((size_t)G4_*KPAD*2);
    bf16_t* WhhB   = (bf16_t*)carve((size_t)G4_*H_*2);
    bf16_t* W0b    = (bf16_t*)carve((size_t)H_*H_*2);
    bf16_t* W1b    = (bf16_t*)carve((size_t)H_*H_*2);
    float*  bsum   = (float*) carve((size_t)G4_*4);
    float*  Xproj  = (float*) carve((size_t)R_*G4_*4);
    float*  hW     = (float*) carve((size_t)B_*G4_*4);
    float*  cst    = (float*) carve((size_t)B_*H_*4);
    bf16_t* hbf    = (bf16_t*)carve((size_t)B_*H_*2);
    bf16_t* hs     = (bf16_t*)carve((size_t)R_*H_*2);
    bf16_t* f1     = (bf16_t*)carve((size_t)R_*H_*2);
    float*  Zbuf   = (float*) carve((size_t)R_*D_*4);
    (void)ws_size;

    // ---- stage 1: paths, payoff, signatures ----
    sde_kernel<<<(B_*D_ + 255)/256, 256, 0, stream>>>(ts, x0, noise, xs, sinc);
    payoff_kernel<<<(B_ + 255)/256, 256, 0, stream>>>(xs, pay);
    sig_kernel<<<R_, 256, 0, stream>>>(xs, ts, txpad);

    // ---- stage 2: the two LSTM+FFN networks ----
    for (int net = 0; net < 2; ++net) {
        const float* Wih = P[net][0]; const float* Whh = P[net][1];
        const float* bih = P[net][2]; const float* bhh = P[net][3];
        const float* W0  = P[net][4]; const float* b0  = P[net][5];
        const float* W1  = P[net][6]; const float* b1  = P[net][7];
        const float* W2  = P[net][8]; const float* b2  = P[net][9];
        const int    O   = (net == 0) ? 1 : D_;

        // weight conversion to bf16 (Wih padded K: 4369 -> 4384)
        {
            size_t tot = (size_t)G4_*KPAD;
            cvt_pad_kernel<<<(unsigned)((tot + 255)/256), 256, 0, stream>>>(Wih, WihB, G4_, KSIG, KPAD);
            tot = (size_t)G4_*H_;
            cvt_pad_kernel<<<(unsigned)((tot + 255)/256), 256, 0, stream>>>(Whh, WhhB, G4_, H_, H_);
            tot = (size_t)H_*H_;
            cvt_pad_kernel<<<(unsigned)((tot + 255)/256), 256, 0, stream>>>(W0, W0b, H_, H_, H_);
            cvt_pad_kernel<<<(unsigned)((tot + 255)/256), 256, 0, stream>>>(W1, W1b, H_, H_, H_);
            bias_sum_kernel<<<(G4_ + 255)/256, 256, 0, stream>>>(bih, bhh, bsum, G4_);
        }

        // Xproj = tx @ Wih^T + (bih+bhh)   [M=11264, N=2048, K=4384]
        {
            dim3 grid(G4_/64, R_/128);
            gemm_bf16<false, true, false, true><<<grid, 256, 0, stream>>>(
                txpad, WihB, bsum, Xproj, (bf16_t*)nullptr, R_, G4_, KPAD);
        }

        // recurrence over W_ timesteps
        init_hc_kernel<<<(B_*H_ + 255)/256, 256, 0, stream>>>(hbf, cst);
        for (int t = 0; t < W_; ++t) {
            dim3 grid(G4_/64, B_/128);
            gemm_bf16<false, true, false, false><<<grid, 256, 0, stream>>>(
                hbf, WhhB, (const float*)nullptr, hW, (bf16_t*)nullptr, B_, G4_, H_);
            lstm_cell_kernel<<<(B_*H_ + 255)/256, 256, 0, stream>>>(
                Xproj, hW, cst, hbf, hs, t);
        }

        // FFN: relu(hs @ W0^T + b0) -> f1 ; relu(f1 @ W1^T + b1) -> hs
        {
            dim3 grid(H_/64, R_/128);
            gemm_bf16<true, false, true, true><<<grid, 256, 0, stream>>>(
                hs, W0b, b0, (float*)nullptr, f1, R_, H_, H_);
            gemm_bf16<true, false, true, true><<<grid, 256, 0, stream>>>(
                f1, W1b, b1, (float*)nullptr, hs, R_, H_, H_);
        }

        // head
        float* hout = (net == 0) ? Ybuf : Zbuf;
        head_kernel<<<(R_*O + 255)/256, 256, 0, stream>>>(hs, W2, b2, hout, O);
    }

    // ---- stage 3: loss ----
    loss_kernel<<<1, 1024, 0, stream>>>(Ybuf, Zbuf, sinc, pay, ts, out);

    (void)in_sizes; (void)n_in; (void)out_size;
}
